// NoNormModel_53128745452235
// MI455X (gfx1250) — matrix operand order repaired and checked by reading, not compiled, not run on hardware
//
#include <hip/hip_runtime.h>

typedef __attribute__((ext_vector_type(16))) _Float16 v16h;
typedef __attribute__((ext_vector_type(8)))  float    v8f;
typedef __attribute__((ext_vector_type(4)))  float    v4f;

#define BB 512
#define TT 1024
#define MMDIM 8
#define AA 32
#define HH 64

// ---- LDS layout (bytes) ----
// Wr f16 swizzled WMMA-B tiles: 256 tiles * 1KB       @ 0
// c state, transposed [64][16] f32                    @ 262144
// cross-wave partials: 16 waves * 32 lanes * 8 f32    @ 266240
// h_last, transposed [64][16] f32                     @ 282624
// xa stage tile [16][32] f16 (bank-swizzled)          @ 286720
// xm stage tile, transposed [8][16] f32               @ 287744
// 0.5*br staged [4096] f32                            @ 288256
#define WR_OFF   0
#define CT_OFF   262144
#define PART_OFF 266240
#define HL_OFF   282624
#define XA_OFF   286720
#define XM_OFF   287744
#define BR_OFF   288256
#define SMEM_BYTES 304640

union Frag { v16h v; v4f f[2]; };

__device__ __forceinline__ float tanh_fast(float x) {
#if __has_builtin(__builtin_amdgcn_tanhf)
    return __builtin_amdgcn_tanhf(x);
#elif __has_builtin(__builtin_amdgcn_tanh_f32)
    return __builtin_amdgcn_tanh_f32(x);
#else
    float e = __builtin_amdgcn_exp2f(-2.0f * x * 1.44269504088896341f);
    return 2.0f * __builtin_amdgcn_rcpf(1.0f + e) - 1.0f;
#endif
}

// sigmoid(x) = 0.5*tanh(0.5*x) + 0.5 ; caller pre-halves the bias.
__device__ __forceinline__ float sigmoid_from_halfarg(float half_x_plus_half_b) {
    return __builtin_fmaf(tanh_fast(half_x_plus_half_b), 0.5f, 0.5f);
}

__global__ __launch_bounds__(512, 1) void nonorm_scan_kernel(
    const float* __restrict__ x_m, const float* __restrict__ x_a,
    const float* __restrict__ Wj,  const float* __restrict__ bj,
    const float* __restrict__ Wr,  const float* __restrict__ br,
    const float* __restrict__ Wo,  const float* __restrict__ bo,
    const float* __restrict__ Wfc, const float* __restrict__ bfc,
    float* __restrict__ out)
{
    extern __shared__ char smem[];
    char*     wr_b = smem + WR_OFF;               // f16 tiles, byte-addressed
    float*    c_s  = (float*)(smem + CT_OFF);     // c_s[h*16 + row]
    float*    part = (float*)(smem + PART_OFF);   // part[w*256 + lane*8 + i]
    float*    hl_s = (float*)(smem + HL_OFF);     // hl_s[h*16 + row]
    char*     xa_b = smem + XA_OFF;               // f16 [16 rows][32 k], swizzled
    float*    xm_t = (float*)(smem + XM_OFF);     // xm_t[m*16 + row]
    float*    br_s = (float*)(smem + BR_OFF);     // 0.5*br, [4096]

    const int tid  = threadIdx.x;
    const int w    = tid >> 5;         // wave 0..15
    const int lane = tid & 31;
    const int kg   = w & 3;            // output hidden group: k in [kg*16, kg*16+16)
    const int hs   = w >> 2;           // source  hidden group: h in [hs*16, hs*16+16)
    const int col  = lane & 15;        // N column inside 16x16 tile
    const int rowb = (lane >> 4) << 3; // batch rows held by this lane: rowb..rowb+7
    const int kB   = lane & 16;        // B-operand: element j holds K = kB + j
    const int b0   = blockIdx.x * 16;
    const int kslot = kg * 16 + col;   // this lane's output hidden index

    // ---- one-time: Wr (f32 row-major 32x4096) -> f16 WMMA-B tiles in LDS,
    //      bank-swizzled (XOR bit4 for lanes with bit3 set -> 2 words/bank).
    for (int idx = tid; idx < AA * HH * HH; idx += 512) {
        int k  = idx >> 12;                 // Wr row (K)    : coalesced over n
        int n  = idx & 4095;                // Wr column (N)
        int rt = n >> 4;                    // tile index (h*4 + kgroup)
        int ln = (n & 15) + (k & 16);       // target lane
        int j  = k & 15;                    // target element
        int boff = ((ln * 32 + j * 2) ^ ((ln & 8) << 1));
        *(_Float16*)(wr_b + rt * 1024 + boff) = (_Float16)Wr[idx];
    }
    // stage half-biases for R gate (consumed with dynamic index in the hi loop)
    for (int idx = tid; idx < HH * HH; idx += 512)
        br_s[idx] = 0.5f * br[idx];
    // zero initial cell state (c_s: 1024 floats)
    c_s[tid] = 0.0f; c_s[tid + 512] = 0.0f;

    // ---- register-resident B fragments + half-biases for J gate (2 tiles/wave)
    Frag bjf[2]; float bj_h[2];
    #pragma unroll
    for (int q = 0; q < 2; ++q) {
        int mm = hs * 2 + q;
        int n  = mm * HH + kslot;
        #pragma unroll
        for (int j = 0; j < 16; ++j)
            bjf[q].v[j] = (_Float16)Wj[(kB + j) * (MMDIM * HH) + n];
        bj_h[q] = 0.5f * bj[n];
    }

    // output-gate fragment (finalize waves w<4 only)
    Frag bof; float bo_h = 0.0f;
    if (w < 4) {
        #pragma unroll
        for (int j = 0; j < 16; ++j)
            bof.v[j] = (_Float16)Wo[(kB + j) * HH + kslot];
        bo_h = 0.5f * bo[kslot];
    }

    // ---- per-thread staging assignments + first-timestep prefetch
    const int r_x  = tid >> 5, k_x = tid & 31;       // xa element (row, k)
    const int rr_x = tid >> 3, m_x = tid & 7;        // xm element (row, m)
    const float* xa_p = x_a + (size_t)(b0 + r_x) * TT * AA + k_x;
    const float* xm_p = x_m + (size_t)(b0 + rr_x) * TT * MMDIM + m_x;
    const int xa_boff = ((r_x * 64 + k_x * 2) ^ ((r_x & 12) << 2)); // swizzled dst
    float xa_reg = xa_p[0];
    float xm_reg = (tid < 128) ? xm_p[0] : 0.0f;

    // A-fragment LDS source addresses (swizzled, two 16B runs)
    const int a_swz  = (col & 12) << 2;
    const int a_off0 = ((col * 64 + ((lane >> 4) << 5)) ^ a_swz);
    // B-tile within-tile offsets (swizzled)
    const int b_off0 = ((lane * 32) ^ ((lane & 8) << 1));

    const int h0 = hs * 16;
    const char*  tb_base = wr_b + (h0 * 4 + kg) * 1024; // first R tile of this wave
    const float* cp_base = c_s + h0 * 16 + rowb;        // first c slice
    const float* bp_base = br_s + h0 * HH + kslot;      // first 0.5*br value

    __syncthreads();

    const v8f zeroC = {};

    for (int t = 0; t < TT; ++t) {
        // ---- publish staged inputs for this step; prefetch next step
        *(_Float16*)(xa_b + xa_boff) = (_Float16)xa_reg;
        if (tid < 128) xm_t[m_x * 16 + rr_x] = xm_reg;
        if (t + 1 < TT) {
            xa_reg = xa_p[(size_t)(t + 1) * AA];
            if (tid < 128) xm_reg = xm_p[(size_t)(t + 1) * MMDIM];
        }
        __syncthreads();

        // ---- A fragment: two 16B vector loads per lane
        Frag af;
        af.f[0] = *(const v4f*)(xa_b + a_off0);
        af.f[1] = *(const v4f*)(xa_b + (a_off0 ^ 16));

        float acc[8];
        #pragma unroll
        for (int i = 0; i < 8; ++i) acc[i] = 0.0f;

        // ---- R gate + m_sys, software-pipelined 1 deep with running pointers:
        //      issue WMMA(h+1) before consuming WMMA(h) -> fills hazard slots.
        Frag bf;
        bf.f[0] = *(const v4f*)(tb_base + b_off0);
        bf.f[1] = *(const v4f*)(tb_base + (b_off0 ^ 16));
        v8f d_cur = __builtin_amdgcn_wmma_f32_16x16x32_f16(
            false, af.v, false, bf.v, (short)0, zeroC, false, false);
        v4f cc0 = *(const v4f*)(cp_base);
        v4f cc1 = *(const v4f*)(cp_base + 4);

        const char*  tb = tb_base;
        const float* cp = cp_base;
        #pragma unroll 3
        for (int hi = 0; hi < 15; ++hi) {
            tb += 4 * 1024;          // next h tile (4 tiles per h, 1KB each)
            cp += 16;                // next c slice
            Frag bn;
            bn.f[0] = *(const v4f*)(tb + b_off0);
            bn.f[1] = *(const v4f*)(tb + (b_off0 ^ 16));
            v8f d_nxt = __builtin_amdgcn_wmma_f32_16x16x32_f16(
                false, af.v, false, bn.v, (short)0, zeroC, false, false);
            v4f nc0 = *(const v4f*)(cp);
            v4f nc1 = *(const v4f*)(cp + 4);
            float brv = bp_base[hi * HH];
            #pragma unroll
            for (int i = 0; i < 8; ++i) {
                float s  = sigmoid_from_halfarg(__builtin_fmaf(d_cur[i], 0.5f, brv));
                float cv = (i < 4) ? cc0[i] : cc1[i - 4];
                acc[i] = __builtin_fmaf(s, cv, acc[i]);
            }
            d_cur = d_nxt; cc0 = nc0; cc1 = nc1;
        }
        {
            float brv = bp_base[15 * HH];
            #pragma unroll
            for (int i = 0; i < 8; ++i) {
                float s  = sigmoid_from_halfarg(__builtin_fmaf(d_cur[i], 0.5f, brv));
                float cv = (i < 4) ? cc0[i] : cc1[i - 4];
                acc[i] = __builtin_fmaf(s, cv, acc[i]);
            }
        }

        // ---- J gate + m_in: issue both WMMAs, then consume
        v8f dj0 = __builtin_amdgcn_wmma_f32_16x16x32_f16(
            false, af.v, false, bjf[0].v, (short)0, zeroC, false, false);
        v8f dj1 = __builtin_amdgcn_wmma_f32_16x16x32_f16(
            false, af.v, false, bjf[1].v, (short)0, zeroC, false, false);
        {
            const float* x0 = xm_t + (hs * 2 + 0) * 16 + rowb;
            const float* x1 = xm_t + (hs * 2 + 1) * 16 + rowb;
            v4f xa0 = *(const v4f*)x0, xa1 = *(const v4f*)(x0 + 4);
            v4f xb0 = *(const v4f*)x1, xb1 = *(const v4f*)(x1 + 4);
            #pragma unroll
            for (int i = 0; i < 8; ++i) {
                float s0 = sigmoid_from_halfarg(__builtin_fmaf(dj0[i], 0.5f, bj_h[0]));
                float s1 = sigmoid_from_halfarg(__builtin_fmaf(dj1[i], 0.5f, bj_h[1]));
                float xv0 = (i < 4) ? xa0[i] : xa1[i - 4];
                float xv1 = (i < 4) ? xb0[i] : xb1[i - 4];
                acc[i] = __builtin_fmaf(s0, xv0, acc[i]);
                acc[i] = __builtin_fmaf(s1, xv1, acc[i]);
            }
        }

        // ---- publish partial m for cross-wave reduction
        {
            v4f p0, p1;
            #pragma unroll
            for (int i = 0; i < 4; ++i) { p0[i] = acc[i]; p1[i] = acc[i + 4]; }
            *(v4f*)(part + w * 256 + lane * 8)     = p0;
            *(v4f*)(part + w * 256 + lane * 8 + 4) = p1;
        }
        __syncthreads();

        // ---- finalize (waves 0..3, one per k-group): o gate, reduce m, update c
        if (w < 4) {
            v8f d = __builtin_amdgcn_wmma_f32_16x16x32_f16(
                false, af.v, false, bof.v, (short)0, zeroC, false, false);
            float msum[8];
            #pragma unroll
            for (int i = 0; i < 8; ++i) msum[i] = 0.0f;
            #pragma unroll
            for (int ss = 0; ss < 4; ++ss) {
                const float* sp = part + (ss * 4 + kg) * 256 + lane * 8;
                v4f p0 = *(const v4f*)sp;
                v4f p1 = *(const v4f*)(sp + 4);
                #pragma unroll
                for (int i = 0; i < 4; ++i) { msum[i] += p0[i]; msum[i + 4] += p1[i]; }
            }
            v4f cn0, cn1, hv0, hv1;
            #pragma unroll
            for (int i = 0; i < 8; ++i) {
                float o  = sigmoid_from_halfarg(__builtin_fmaf(d[i], 0.5f, bo_h));
                float m  = msum[i];
                float cn = (1.0f - o) * m;
                float hv = o * m;
                if (i < 4) { cn0[i] = cn; hv0[i] = hv; }
                else       { cn1[i - 4] = cn; hv1[i - 4] = hv; }
            }
            *(v4f*)(c_s + kslot * 16 + rowb)     = cn0;
            *(v4f*)(c_s + kslot * 16 + rowb + 4) = cn1;
            if (t == TT - 1) {
                *(v4f*)(hl_s + kslot * 16 + rowb)     = hv0;
                *(v4f*)(hl_s + kslot * 16 + rowb + 4) = hv1;
            }
        }
        // NOTE: no barrier here — next iteration's post-staging barrier orders
        // finalize's c_s writes against the next step's reads (staging touches
        // only xa/xm buffers, which step-t readers finished before the publish
        // barrier above).
    }
    __syncthreads();

    // ---- epilogue: c_final (B,H) at offset 512, readout (B,1) at offset 0
    for (int idx = tid; idx < 16 * HH; idx += 512) {
        int r = idx & 15, h = idx >> 4;
        out[BB + (size_t)(b0 + r) * HH + h] = c_s[h * 16 + r];
    }
    if (w == 0 && lane < 16) {
        float s = bfc[0];
        for (int h = 0; h < HH; ++h)
            s = __builtin_fmaf(hl_s[h * 16 + lane], Wfc[h], s);
        out[b0 + lane] = s;
    }
}

extern "C" void kernel_launch(void* const* d_in, const int* in_sizes, int n_in,
                              void* d_out, int out_size, void* d_ws, size_t ws_size,
                              hipStream_t stream) {
    (void)in_sizes; (void)n_in; (void)out_size; (void)d_ws; (void)ws_size;
    const float* x_m = (const float*)d_in[0];
    const float* x_a = (const float*)d_in[1];
    const float* Wj  = (const float*)d_in[2];
    const float* bj  = (const float*)d_in[3];
    const float* Wr  = (const float*)d_in[4];
    const float* br  = (const float*)d_in[5];
    const float* Wo  = (const float*)d_in[6];
    const float* bo  = (const float*)d_in[7];
    const float* Wfc = (const float*)d_in[8];
    const float* bfc = (const float*)d_in[9];
    float* out = (float*)d_out;

    // ~298 KB dynamic LDS per workgroup (CDNA5 WGP has 320 KB) — raise the cap.
    hipFuncSetAttribute((const void*)nonorm_scan_kernel,
                        hipFuncAttributeMaxDynamicSharedMemorySize, SMEM_BYTES);

    nonorm_scan_kernel<<<dim3(BB / 16), dim3(512), SMEM_BYTES, stream>>>(
        x_m, x_a, Wj, bj, Wr, br, Wo, bo, Wfc, bfc, out);
}
